// Model_79731772882946
// MI455X (gfx1250) — matrix operand order repaired and checked by reading, not compiled, not run on hardware
//
#include <hip/hip_runtime.h>
#include <math.h>

// ---------- types ----------
typedef __attribute__((ext_vector_type(16))) __bf16        v16bf;
typedef __attribute__((ext_vector_type(8)))  float         v8f;
typedef __attribute__((ext_vector_type(4)))  unsigned int  u32x4;

union Frag { u32x4 q[2]; v16bf v; unsigned short s[16]; };

__device__ __forceinline__ unsigned short f2bf(float f) {
  union { float f; unsigned u; } v; v.f = f;
  unsigned r = v.u + 0x7FFFu + ((v.u >> 16) & 1u);
  return (unsigned short)(r >> 16);
}
__device__ __forceinline__ float bf2f(unsigned short h) {
  union { unsigned u; float f; } v; v.u = ((unsigned)h) << 16;
  return v.f;
}
__device__ __forceinline__ v8f z8() {
  v8f z = {0.f,0.f,0.f,0.f,0.f,0.f,0.f,0.f}; return z;
}
__device__ __forceinline__ float sigm(float x) { return 1.f / (1.f + __expf(-x)); }

// ---------- CDNA5 async global->LDS copy (16B per lane) ----------
// GLOBAL_LOAD_ASYNC_TO_LDS_B128 (ISA §10.4 / §15.18): VDST = per-lane LDS byte
// address, VADDR = 64-bit global address. Tracked by ASYNCcnt.
__device__ __forceinline__ void async_g2l_b128(const void* gsrc, void* lds) {
  unsigned lo = (unsigned)(unsigned long long)lds;   // low 32 bits = LDS byte offset
  asm volatile("global_load_async_to_lds_b128 %0, %1, off"
               :: "v"(lo), "v"(gsrc) : "memory");
}
__device__ __forceinline__ void wait_async0() {
  asm volatile("s_wait_asynccnt 0" ::: "memory");
}

// ---------- problem constants ----------
#define VSZ 100000
#define EDIM 256
#define HDIM 256
#define BB 64
#define SS 64
#define NN 4096          // BB*SS
#define KK 8

// ---------- kernel: fp32 -> bf16 convert ----------
__global__ void conv_bf16(const float* __restrict__ s, unsigned short* __restrict__ d, size_t n) {
  for (size_t i = (size_t)blockIdx.x * blockDim.x + threadIdx.x; i < n;
       i += (size_t)gridDim.x * blockDim.x)
    d[i] = f2bf(s[i]);
}

// ---------- kernel: pack W[N,K] row-major fp32 -> CDNA5 B-fragment bf16 ----------
// frag[((nt*KT + kt)*32 + lane)*16 + i] = B[k][n] = W[n][k]
//   with k = kt*32 + 16*(lane>>4) + i, n = nt*16 + (lane&15)
__global__ void pack_b_frag(const float* __restrict__ W, unsigned short* __restrict__ dst,
                            int K, int N) {
  const int KT = K >> 5;
  const int total = (N >> 4) * KT * 512;
  for (int idx = blockIdx.x * blockDim.x + threadIdx.x; idx < total;
       idx += gridDim.x * blockDim.x) {
    int i    = idx & 15;
    int lane = (idx >> 4) & 31;
    int t    = idx >> 9;                 // nt*KT + kt
    int kt   = t % KT, nt = t / KT;
    int k = kt * 32 + ((lane >> 4) << 4) + i;
    int n = nt * 16 + (lane & 15);
    dst[idx] = f2bf(W[(size_t)n * K + k]);
  }
}

// ---------- kernel: WMMA GEMM  C[M,N] = A[M,K](bf16) x Bfrag + bias ----------
// 256 threads / 8 waves per block. Block owns one 16-row A tile (async-staged
// into LDS, shared by all 8 waves) and 8 consecutive N-tiles (one per wave).
__global__ void __launch_bounds__(256)
gemm_bf16_wmma(const unsigned short* __restrict__ A,
               const unsigned short* __restrict__ Bfrag,
               const float* __restrict__ bias,
               float* __restrict__ outF,
               unsigned short* __restrict__ outB,
               int M, int K, int N) {
  __shared__ unsigned short Atile[16 * 512];       // up to K=512 -> 16KB
  const int tilesN = N >> 4;
  const int NG     = tilesN >> 3;                  // groups of 8 n-tiles
  const int KT     = K >> 5;
  const int tid  = threadIdx.x;
  const int ng   = blockIdx.x % NG;
  const int mt   = blockIdx.x / NG;
  const int wv   = tid >> 5, lane = tid & 31;
  const int l15  = lane & 15, lh = lane >> 4;

  // stage A tile [16 x K] into LDS with async DMA (16B chunks, coalesced)
  const int cpr = K >> 3;                          // 16B chunks per row
  for (int c = tid; c < 16 * cpr; c += 256) {
    int r  = c / cpr, cc = c - r * cpr;
    int ar = mt * 16 + r; if (ar >= M) ar = M - 1;
    async_g2l_b128(A + (size_t)ar * K + cc * 8, Atile + c * 8);
  }
  wait_async0();
  __syncthreads();

  const int nt = ng * 8 + wv;
  const unsigned short* Bp = Bfrag + ((size_t)nt * KT * 32 + lane) * 16;

  v8f acc = z8();
  for (int kt = 0; kt < KT; ++kt) {
    Frag a, b;
    const unsigned short* ap = Atile + l15 * K + kt * 32 + 16 * lh;
    a.q[0] = *(const u32x4*)ap;
    a.q[1] = *(const u32x4*)(ap + 8);
    b.q[0] = *(const u32x4*)Bp;
    b.q[1] = *(const u32x4*)(Bp + 8);
    __builtin_prefetch(Bp + 512, 0, 1);
    Bp += 512;
    acc = __builtin_amdgcn_wmma_f32_16x16x32_bf16(false, a.v, false, b.v,
                                                  (short)0, acc, false, false);
  }
  const int col = nt * 16 + l15;
  const float bv = bias ? bias[col] : 0.f;
#pragma unroll
  for (int v = 0; v < 8; ++v) {
    int row = mt * 16 + lh * 8 + v;
    if (row < M) {
      float val = acc[v] + bv;
      if (outB) outB[(size_t)row * N + col] = f2bf(val);
      else      outF[(size_t)row * N + col] = val;
    }
  }
}

// ---------- kernel: per-sample tree reduce (sum-into-parent + amax incl. 0) ----------
__global__ void tree_reduce(const int* __restrict__ t0, const int* __restrict__ t1,
                            const int* __restrict__ t2,
                            const unsigned short* __restrict__ Yt,
                            unsigned short* __restrict__ nodeOut, int encOffRows) {
  const int n = blockIdx.x;        // 0..4095
  const int e = threadIdx.x;       // 0..255
  float c0 = bf2f(Yt[(size_t)t0[n] * EDIM + e]);
  float mx = 0.f;                  // include_self zeros init
#pragma unroll 1
  for (int k = 0; k < KK; ++k) {
    float c1 = bf2f(Yt[(size_t)t1[n * KK + k] * EDIM + e]);
#pragma unroll 1
    for (int j = 0; j < KK; ++j) {
      float y2 = bf2f(Yt[(size_t)t2[(n * KK + k) * KK + j] * EDIM + e]);
      c1 += y2;
      mx = fmaxf(mx, y2);
    }
    mx = fmaxf(mx, c1);
    c0 += c1;
  }
  mx = fmaxf(mx, c0);
  const int s = n & (SS - 1), b = n >> 6;   // n = b*S + s
  nodeOut[((size_t)(encOffRows + s * BB + b)) * EDIM + e] = f2bf(mx);
}

// ---------- kernel: persistent per-(enc,dir) GRU recurrence ----------
// gi:   [dir][8192][768] fp32 (already includes bih); rows = enc*4096 + s*64 + b
// Whh:  fragment-packed per dir (K=256, N=768), stride 48*8*512 halves
// hout: [enc*4096 + s*64 + b][512] bf16, cols dir*256 + j
__global__ void __launch_bounds__(512)
gru_recurrence(const float* __restrict__ gi,
               const unsigned short* __restrict__ WhhFrag,
               const float* __restrict__ bhh,
               unsigned short* __restrict__ hout) {
  __shared__ float          hF[BB * HDIM];   // 64 KB
  __shared__ unsigned short hB[BB * HDIM];   // 32 KB
  const int enc = blockIdx.x >> 1, dir = blockIdx.x & 1;
  const int tid = threadIdx.x, wv = tid >> 5, lane = tid & 31;
  const int l15 = lane & 15, lh = lane >> 4;
  const int jt = wv;                 // 16 waves: wave owns h-columns jt*16..+15
  const int j  = jt * 16 + l15;

  const float* giD = gi + ((size_t)dir * 2 * NN + (size_t)enc * NN) * 768;
  const unsigned short* Bb = WhhFrag + (size_t)dir * (48 * 8 * 512);
  const float br = bhh[dir * 768 + j];
  const float bz = bhh[dir * 768 + 256 + j];
  const float bn = bhh[dir * 768 + 512 + j];

  for (int i = tid; i < BB * HDIM; i += 512) { hF[i] = 0.f; hB[i] = 0; }
  __syncthreads();

  for (int t = 0; t < SS; ++t) {
    const int src = dir ? (SS - 1 - t) : t;
    float hnew[4][8];
#pragma unroll 1
    for (int mt4 = 0; mt4 < 4; ++mt4) {        // 4 M-tiles of 16 rows
      v8f accR = z8(), accZ = z8(), accN = z8();
#pragma unroll
      for (int kt = 0; kt < 8; ++kt) {         // K = 256
        Frag a;
        const int m = mt4 * 16 + l15;
        const unsigned short* hp = hB + m * HDIM + kt * 32 + 16 * lh;
        a.q[0] = *(const u32x4*)hp;
        a.q[1] = *(const u32x4*)(hp + 8);
        // r gate (n-tile jt)
        {
          Frag b; const unsigned short* bp = Bb + ((size_t)((0 * 16 + jt) * 8 + kt) * 32 + lane) * 16;
          b.q[0] = *(const u32x4*)bp; b.q[1] = *(const u32x4*)(bp + 8);
          accR = __builtin_amdgcn_wmma_f32_16x16x32_bf16(false, a.v, false, b.v, (short)0, accR, false, false);
        }
        // z gate (n-tile 16+jt)
        {
          Frag b; const unsigned short* bp = Bb + ((size_t)((1 * 16 + jt) * 8 + kt) * 32 + lane) * 16;
          b.q[0] = *(const u32x4*)bp; b.q[1] = *(const u32x4*)(bp + 8);
          accZ = __builtin_amdgcn_wmma_f32_16x16x32_bf16(false, a.v, false, b.v, (short)0, accZ, false, false);
        }
        // n gate (n-tile 32+jt)
        {
          Frag b; const unsigned short* bp = Bb + ((size_t)((2 * 16 + jt) * 8 + kt) * 32 + lane) * 16;
          b.q[0] = *(const u32x4*)bp; b.q[1] = *(const u32x4*)(bp + 8);
          accN = __builtin_amdgcn_wmma_f32_16x16x32_bf16(false, a.v, false, b.v, (short)0, accN, false, false);
        }
      }
      // fused gates -> hnew (kept in registers; LDS writes deferred past barrier)
#pragma unroll
      for (int v = 0; v < 8; ++v) {
        const int m = mt4 * 16 + lh * 8 + v;
        const float* gr = giD + (size_t)(src * BB + m) * 768;
        float r = sigm(gr[j]       + accR[v] + br);
        float z = sigm(gr[256 + j] + accZ[v] + bz);
        float nn = tanhf(gr[512 + j] + r * (accN[v] + bn));
        hnew[mt4][v] = (1.f - z) * nn + z * hF[m * HDIM + j];
      }
    }
    __syncthreads();   // all waves done reading hB for this step
#pragma unroll
    for (int q = 0; q < 4; ++q)
#pragma unroll
      for (int v = 0; v < 8; ++v) {
        const int m = q * 16 + lh * 8 + v;
        float hv = hnew[q][v];
        hF[m * HDIM + j] = hv;
        unsigned short hb = f2bf(hv);
        hB[m * HDIM + j] = hb;
        hout[((size_t)enc * NN + src * BB + m) * 512 + dir * 256 + j] = hb;
      }
    __syncthreads();   // writes visible before next step's fragment loads
  }
}

// ---------- kernel: final reductions ----------
// out[b] = dot(max_s h1, max_s h2) + (sum_s h1 * sum_s h2) . lin2_w + lin2_b
__global__ void finalize(const float* __restrict__ hc, const float* __restrict__ lin2_w,
                         const float* __restrict__ lin2_b, float* __restrict__ out) {
  __shared__ float red[256];
  const int b = blockIdx.x, e = threadIdx.x;
  float m1 = -3.4e38f, m2 = -3.4e38f, s1 = 0.f, s2 = 0.f;
  for (int s = 0; s < SS; ++s) {
    float h1 = hc[((size_t)(s * BB + b)) * HDIM + e];
    float h2 = hc[((size_t)(NN + s * BB + b)) * HDIM + e];
    m1 = fmaxf(m1, h1); m2 = fmaxf(m2, h2); s1 += h1; s2 += h2;
  }
  red[e] = m1 * m2 + (s1 * s2) * lin2_w[e];
  __syncthreads();
  for (int off = 128; off > 0; off >>= 1) {
    if (e < off) red[e] += red[e + off];
    __syncthreads();
  }
  if (e == 0) out[b] = red[0] + lin2_b[0];
}

// ---------- workspace layout (bytes) ----------
#define OFF_EMBED 0ULL                       // bf16 V*E          = 51,200,000
#define OFF_YT    51200000ULL                // bf16 V*256        = 51,200,000
#define OFF_GI    0ULL                       // fp32 [2][8192][768] = 50,331,648 (aliases embed; embed dead by then)
#define OFF_W     102400000ULL               // packed weight fragments
#define OFF_NODE  106725376ULL               // bf16 [2][4096][256]
#define OFF_HL0   110919680ULL               // bf16 [2][4096][512]
#define OFF_HL1   119308288ULL
#define OFF_HC    127696896ULL               // fp32 [2][4096][256]
#define WS_NEED   136085504ULL

extern "C" void kernel_launch(void* const* d_in, const int* in_sizes, int n_in,
                              void* d_out, int out_size, void* d_ws, size_t ws_size,
                              hipStream_t stream) {
  (void)in_sizes; (void)n_in; (void)out_size;
  if (ws_size < WS_NEED) return;

  const int*   x1t0   = (const int*)d_in[0];
  const int*   x1t1   = (const int*)d_in[1];
  const int*   x1t2   = (const int*)d_in[2];
  const int*   x2t0   = (const int*)d_in[3];
  const int*   x2t1   = (const int*)d_in[4];
  const int*   x2t2   = (const int*)d_in[5];
  const float* embedW = (const float*)d_in[6];
  const float* linW   = (const float*)d_in[7];
  const float* linB   = (const float*)d_in[8];
  const float* Wih0   = (const float*)d_in[9];
  const float* Whh0   = (const float*)d_in[10];
  const float* bih0   = (const float*)d_in[11];
  const float* bhh0   = (const float*)d_in[12];
  const float* Wih1   = (const float*)d_in[13];
  const float* Whh1   = (const float*)d_in[14];
  const float* bih1   = (const float*)d_in[15];
  const float* bhh1   = (const float*)d_in[16];
  const float* combW  = (const float*)d_in[17];
  const float* combB  = (const float*)d_in[18];
  const float* lin2W  = (const float*)d_in[19];
  const float* lin2B  = (const float*)d_in[20];

  char* ws = (char*)d_ws;
  unsigned short* embB  = (unsigned short*)(ws + OFF_EMBED);
  unsigned short* Yt    = (unsigned short*)(ws + OFF_YT);
  float*          gi    = (float*)(ws + OFF_GI);
  unsigned short* linF  = (unsigned short*)(ws + OFF_W);
  unsigned short* wih0F = linF  + 65536;     // lin frag: 65536 halves
  unsigned short* whh0F = wih0F + 393216;    // Wih0: 2*196608
  unsigned short* wih1F = whh0F + 393216;    // Whh0: 2*196608
  unsigned short* whh1F = wih1F + 786432;    // Wih1: 2*393216
  unsigned short* combF = whh1F + 393216;    // Whh1: 2*196608
  unsigned short* nodeB = (unsigned short*)(ws + OFF_NODE);
  unsigned short* hl0   = (unsigned short*)(ws + OFF_HL0);
  unsigned short* hl1   = (unsigned short*)(ws + OFF_HL1);
  float*          hc    = (float*)(ws + OFF_HC);

  // 1) convert embedding table to bf16; pack all weight B-matrices to fragments
  conv_bf16<<<2048, 256, 0, stream>>>(embedW, embB, (size_t)VSZ * EDIM);
  pack_b_frag<<<256, 256, 0, stream>>>(linW, linF, 256, 256);
  for (int d2 = 0; d2 < 2; ++d2) {
    pack_b_frag<<<768,  256, 0, stream>>>(Wih0 + (size_t)d2 * 768 * 256, wih0F + d2 * 196608, 256, 768);
    pack_b_frag<<<768,  256, 0, stream>>>(Whh0 + (size_t)d2 * 768 * 256, whh0F + d2 * 196608, 256, 768);
    pack_b_frag<<<1536, 256, 0, stream>>>(Wih1 + (size_t)d2 * 768 * 512, wih1F + d2 * 393216, 512, 768);
    pack_b_frag<<<768,  256, 0, stream>>>(Whh1 + (size_t)d2 * 768 * 256, whh1F + d2 * 196608, 256, 768);
  }
  pack_b_frag<<<512, 256, 0, stream>>>(combW, combF, 512, 256);

  // 2) Ytable = embed @ lin_w^T + lin_b  for the whole vocab (bf16 out, L2-resident)
  //    grid = tilesM * (tilesN/8) = 6250 * 2
  gemm_bf16_wmma<<<12500, 256, 0, stream>>>(embB, linF, linB, nullptr, Yt, VSZ, 256, 256);

  // 3) per-sample tree reduce -> node x [2][S][B][256] bf16
  tree_reduce<<<NN, 256, 0, stream>>>(x1t0, x1t1, x1t2, Yt, nodeB, 0);
  tree_reduce<<<NN, 256, 0, stream>>>(x2t0, x2t1, x2t2, Yt, nodeB, NN);

  // 4) layer-0 input gates gi = x @ Wih0^T + bih0 (per direction), then recurrence
  //    grid = 512 * (48/8) = 3072
  for (int d2 = 0; d2 < 2; ++d2)
    gemm_bf16_wmma<<<3072, 256, 0, stream>>>(nodeB, wih0F + d2 * 196608, bih0 + d2 * 768,
                                             gi + (size_t)d2 * 6291456, nullptr, 8192, 256, 768);
  gru_recurrence<<<4, 512, 0, stream>>>(gi, whh0F, bhh0, hl0);

  // 5) layer-1 input gates from layer-0 output, then recurrence
  for (int d2 = 0; d2 < 2; ++d2)
    gemm_bf16_wmma<<<3072, 256, 0, stream>>>(hl0, wih1F + d2 * 393216, bih1 + d2 * 768,
                                             gi + (size_t)d2 * 6291456, nullptr, 8192, 512, 768);
  gru_recurrence<<<4, 512, 0, stream>>>(gi, whh1F, bhh1, hl1);

  // 6) combine linear: h = h2 @ comb_w^T + comb_b  (fp32 out); grid = 512 * 2
  gemm_bf16_wmma<<<1024, 256, 0, stream>>>(hl1, combF, combB, hc, nullptr, 8192, 512, 256);

  // 7) final similarity reductions -> out[64]
  finalize<<<BB, 256, 0, stream>>>(hc, lin2W, lin2B, (float*)d_out);
}
